// Attention_43473658970578
// MI455X (gfx1250) — matrix operand order repaired and checked by reading, not compiled, not run on hardware
//
#include <hip/hip_runtime.h>
#include <math.h>

typedef __attribute__((ext_vector_type(16))) _Float16 v16h;
typedef __attribute__((ext_vector_type(8)))  float    v8f;

#define Bdim 2
#define Hdim 16
#define Sdim 2048
#define Ddim 64
#define QT   16                   // query rows per workgroup (one WMMA M-tile)
#define NW   4                    // waves per workgroup (wave32)
#define SMEM_BYTES (QT * Sdim * 4 + 2 * QT * 4)   // 128KB scores + row max/sum

__global__ __launch_bounds__(NW * 32)
void attn_fused_kernel(const float* __restrict__ Q,
                       const float* __restrict__ K,
                       const float* __restrict__ V,
                       const int*   __restrict__ M,
                       float* __restrict__ out,
                       float* __restrict__ pattn)
{
    extern __shared__ float smem[];
    float* sc   = smem;               // [QT][Sdim] scores, later exp values
    float* rmax = smem + QT * Sdim;   // [QT]
    float* rsum = rmax + QT;          // [QT]

    const int tid  = threadIdx.x;
    const int lane = tid & 31;
    const int wave = tid >> 5;
    const int nlo  = lane & 15;       // N / row-in-tile index
    const int hi   = lane >> 4;       // half-wave select

    const int qt = blockIdx.x, h = blockIdx.y, b = blockIdx.z;
    const int qbase = qt * QT;
    const size_t bh = (size_t)b * Hdim + h;
    const float* Qb = Q + bh * Sdim * Ddim;
    const float* Kb = K + bh * Sdim * Ddim;
    const float* Vb = V + bh * Sdim * Ddim;
    const int*   Mb = M + (size_t)b * Sdim * Sdim;

    // ---- Q tile in WMMA f16 A-layout (16x32 per step, two steps cover D=64) ----
    // lane L: row=M=nlo, elems 0..15 -> K=hi*16..hi*16+15
    v16h aQ[2];
    {
        const float* qrow = Qb + (size_t)(qbase + nlo) * Ddim;
        const int off = hi * 16;
        #pragma unroll
        for (int w = 0; w < 2; ++w) {
            #pragma unroll
            for (int e = 0; e < 16; ++e) aQ[w][e] = (_Float16)qrow[w * 32 + off + e];
        }
    }

    // ---- Pass 1: scores = mask ? QK^T * 0.125 : -1e9  -> LDS ----
    for (int kb = wave * 16; kb < Sdim; kb += NW * 16) {
        // B-layout (32x16): lane holds column n=nlo; low half K=0..15, high K=16..31
        v16h bK[2];
        const float* krow = Kb + (size_t)(kb + nlo) * Ddim;
        const int dbase = hi * 16;
        #pragma unroll
        for (int w = 0; w < 2; ++w) {
            #pragma unroll
            for (int e = 0; e < 16; ++e)
                bK[w][e] = (_Float16)krow[w * 32 + dbase + e];
        }

        v8f c = {};
        c = __builtin_amdgcn_wmma_f32_16x16x32_f16(false, aQ[0], false, bK[0], (short)0, c, false, false);
        c = __builtin_amdgcn_wmma_f32_16x16x32_f16(false, aQ[1], false, bK[1], (short)0, c, false, false);

        const int col = kb + nlo;
        #pragma unroll
        for (int j = 0; j < 8; ++j) {           // C/D layout: VGPR j -> row j (+8 for high half)
            const int row = j + hi * 8;
            const int mv  = Mb[(size_t)(qbase + row) * Sdim + col];
            sc[row * Sdim + col] = (mv == 0) ? -1e9f : c[j] * 0.125f;
        }
    }
    __syncthreads();

    // ---- softmax: 8 threads per row, 256 elems each, shuffle reductions ----
    const int srow = tid >> 3;
    const int sch  = tid & 7;
    float* rp = sc + srow * Sdim + sch * 256;

    float lmax = -INFINITY;
    for (int i = 0; i < 256; i += 4) {
        float4 v = *(const float4*)(rp + i);
        lmax = fmaxf(lmax, fmaxf(fmaxf(v.x, v.y), fmaxf(v.z, v.w)));
    }
    #pragma unroll
    for (int o = 4; o > 0; o >>= 1) lmax = fmaxf(lmax, __shfl_down(lmax, o, 8));
    if (sch == 0) rmax[srow] = lmax;
    __syncthreads();

    const float rm = rmax[srow];
    float lsum = 0.f;
    for (int i = 0; i < 256; i += 4) {
        float4 v = *(const float4*)(rp + i);
        v.x = __expf(v.x - rm); v.y = __expf(v.y - rm);
        v.z = __expf(v.z - rm); v.w = __expf(v.w - rm);
        *(float4*)(rp + i) = v;                 // LDS now holds e = exp(s - max)
        lsum += (v.x + v.y) + (v.z + v.w);
    }
    #pragma unroll
    for (int o = 4; o > 0; o >>= 1) lsum += __shfl_down(lsum, o, 8);
    if (sch == 0) rsum[srow] = lsum;
    __syncthreads();

    // ---- stream p_attn = e * (1/sum) to HBM (the dominant 537MB write) ----
    {
        const float inv = 1.0f / rsum[srow];
        float* prow = pattn + (bh * Sdim + (size_t)(qbase + srow)) * Sdim + sch * 256;
        for (int i = 0; i < 256; i += 4) {
            float4 v = *(const float4*)(rp + i);
            v.x *= inv; v.y *= inv; v.z *= inv; v.w *= inv;
            *(float4*)(prow + i) = v;
        }
    }

    // ---- Pass 2: out = (e . V) * inv_sum ; wave owns D-slice [wave*16, wave*16+16) ----
    const int d2 = wave * 16;
    v8f acc = {};
    for (int kt = 0; kt < Sdim; kt += 32) {
        v16h aP;                                 // A-layout from LDS e-values
        {
            const float* p0 = sc + nlo * Sdim + kt;
            const int off = hi * 16;
            #pragma unroll
            for (int e = 0; e < 16; ++e) aP[e] = (_Float16)p0[off + e];
        }
        v16h bV;                                 // B-layout: lane col d2+nlo, K rows kt+hi*16+e
        {
            const float* v0 = Vb + (size_t)(kt + hi * 16) * Ddim + d2 + nlo;
            #pragma unroll
            for (int e = 0; e < 16; ++e) bV[e] = (_Float16)v0[e * Ddim];
        }
        acc = __builtin_amdgcn_wmma_f32_16x16x32_f16(false, aP, false, bV, (short)0, acc, false, false);
    }
    #pragma unroll
    for (int j = 0; j < 8; ++j) {
        const int row = j + hi * 8;
        out[(bh * Sdim + (size_t)(qbase + row)) * Ddim + d2 + nlo] = acc[j] * (1.0f / rsum[row]);
    }
}

extern "C" void kernel_launch(void* const* d_in, const int* in_sizes, int n_in,
                              void* d_out, int out_size, void* d_ws, size_t ws_size,
                              hipStream_t stream) {
    (void)in_sizes; (void)n_in; (void)out_size; (void)d_ws; (void)ws_size;
    const float* Q = (const float*)d_in[0];
    const float* K = (const float*)d_in[1];
    const float* V = (const float*)d_in[2];
    const int*   M = (const int*)d_in[3];
    float* out   = (float*)d_out;                                   // [B,H,S,D]
    float* pattn = out + (size_t)Bdim * Hdim * Sdim * Ddim;         // [B,H,S,S]

    (void)hipFuncSetAttribute((const void*)attn_fused_kernel,
                              hipFuncAttributeMaxDynamicSharedMemorySize, SMEM_BYTES);

    dim3 grid(Sdim / QT, Hdim, Bdim);   // 128 x 16 x 2 workgroups
    dim3 block(NW * 32);                // 4 wave32 waves
    attn_fused_kernel<<<grid, block, SMEM_BYTES, stream>>>(Q, K, V, M, out, pattn);
}
